// SelfAttention_7730941132970
// MI455X (gfx1250) — matrix operand order repaired and checked by reading, not compiled, not run on hardware
//
#include <hip/hip_runtime.h>

#define B_   8
#define HW   4096      // N = H*W
#define C_   512
#define DQK  64
#define NCOLS 640      // 64 q + 64 k + 512 v

typedef __bf16 bf16_t;
typedef __attribute__((ext_vector_type(16))) __bf16 v16bf;
typedef __attribute__((ext_vector_type(8)))  __bf16 v8bf;
typedef __attribute__((ext_vector_type(8)))  float  v8f;

union BF16x16 { v16bf v; v8bf h[2]; };

__device__ inline v8f zero8() { v8f z = {0.f,0.f,0.f,0.f,0.f,0.f,0.f,0.f}; return z; }

__device__ inline v8f wmma_bf16(v16bf a, v16bf b, v8f c) {
    return __builtin_amdgcn_wmma_f32_16x16x32_bf16(false, a, false, b, (short)0, c, false, false);
}

// A fragment: 16x32 bf16, M rows row0..row0+15 of row-major [*, ld], K = k0..k0+31.
// lane&15 = M row; lanes 0-15 hold K {0..7,16..23}; lanes 16-31 hold K {8..15,24..31}.
__device__ inline v16bf load_frag_a(const bf16_t* base, int ld, int row0, int k0, int lane) {
    int m = lane & 15, hi = lane >> 4;
    const bf16_t* p = base + (size_t)(row0 + m) * ld + k0 + 8 * hi;
    BF16x16 f;
    f.h[0] = *(const v8bf*)p;
    f.h[1] = *(const v8bf*)(p + 16);
    return f.v;
}

// B fragment: 32x16 bf16 (KxN) read from B^T stored row-major [Ncols, ld(K)].
// lane&15 = N col; lanes 0-15 hold K {0..7,16..23}; lanes 16-31 hold K {8..15,24..31}.
__device__ inline v16bf load_frag_b(const bf16_t* baseT, int ld, int col0, int k0, int lane) {
    int n = lane & 15, hi = lane >> 4;
    const bf16_t* p = baseT + (size_t)(col0 + n) * ld + k0 + 8 * hi;
    BF16x16 f;
    f.h[0] = *(const v8bf*)p;
    f.h[1] = *(const v8bf*)(p + 16);
    return f.v;
}

// ---------------- Kernel 1: pack to bf16, transpose weights, concat biases ----
__global__ void convert_pack_kernel(const float* __restrict__ x,
                                    const float* __restrict__ Wq,
                                    const float* __restrict__ Wk,
                                    const float* __restrict__ Wv,
                                    const float* __restrict__ bq,
                                    const float* __restrict__ bk,
                                    const float* __restrict__ bv,
                                    bf16_t* __restrict__ xb,
                                    bf16_t* __restrict__ wT,
                                    float* __restrict__ bcat) {
    int idx = blockIdx.x * blockDim.x + threadIdx.x;
    if (idx < B_ * HW * C_) xb[idx] = (bf16_t)x[idx];
    if (idx < NCOLS * C_) {
        int n = idx / C_, k = idx % C_;
        float w = (n < DQK)     ? Wq[k * DQK + n]
                : (n < 2 * DQK) ? Wk[k * DQK + (n - DQK)]
                                : Wv[k * C_ + (n - 2 * DQK)];
        wT[n * C_ + k] = (bf16_t)w;   // store W transposed: [NCOLS, C]
    }
    if (idx < NCOLS) {
        bcat[idx] = (idx < DQK) ? bq[idx]
                  : (idx < 2 * DQK) ? bk[idx - DQK]
                                    : bv[idx - 2 * DQK];
    }
}

// ---------------- Kernel 2: fused QKV projection GEMM (bf16 WMMA) ------------
// out: qk row-major [B*N, 128] (cols 0..63 = q, 64..127 = k), v transposed [B, C, N]
__global__ __launch_bounds__(256) void qkv_gemm_kernel(const bf16_t* __restrict__ xb,
                                                       const bf16_t* __restrict__ wT,
                                                       const float* __restrict__ bcat,
                                                       bf16_t* __restrict__ qk,
                                                       bf16_t* __restrict__ vT) {
    int lane = threadIdx.x & 31;
    int wid  = blockIdx.x * 8 + (threadIdx.x >> 5);   // 20480 wave tiles
    int rt = wid / 10, ct = wid % 10;
    int row0 = rt * 16, col0 = ct * 64;
    int m = lane & 15, hi = lane >> 4;

    v8f acc[4];
    #pragma unroll
    for (int t = 0; t < 4; ++t) acc[t] = zero8();

    for (int k0 = 0; k0 < C_; k0 += 32) {
        v16bf a = load_frag_a(xb, C_, row0, k0, lane);
        #pragma unroll
        for (int t = 0; t < 4; ++t) {
            v16bf b = load_frag_b(wT, C_, col0 + t * 16, k0, lane);
            acc[t] = wmma_bf16(a, b, acc[t]);
        }
    }
    #pragma unroll
    for (int t = 0; t < 4; ++t) {
        int col = col0 + t * 16 + m;
        float bias = bcat[col];
        #pragma unroll
        for (int r = 0; r < 8; ++r) {
            int row = row0 + r + 8 * hi;
            bf16_t val = (bf16_t)(acc[t][r] + bias);
            if (col < 2 * DQK) {
                qk[(size_t)row * (2 * DQK) + col] = val;
            } else {
                int bb = row >> 12, i = row & (HW - 1);
                vT[((size_t)bb * C_ + (col - 2 * DQK)) * HW + i] = val;
            }
        }
    }
}

// ---------------- Kernel 3: softmax row stats m_i, 1/l_i (flash pass 1) ------
__global__ __launch_bounds__(256) void rowstats_kernel(const bf16_t* __restrict__ qk,
                                                       float* __restrict__ mrow,
                                                       float* __restrict__ linv) {
    int lane = threadIdx.x & 31;
    int wid  = blockIdx.x * 8 + (threadIdx.x >> 5);   // 2048 waves
    int b  = wid >> 8;
    int i0 = (wid & 255) * 16;
    int m = lane & 15, hi = lane >> 4;

    v16bf aq0 = load_frag_a(qk, 128, b * HW + i0, 0, lane);
    v16bf aq1 = load_frag_a(qk, 128, b * HW + i0, 32, lane);

    float mr[8], lr[8];
    #pragma unroll
    for (int r = 0; r < 8; ++r) { mr[r] = -__builtin_inff(); lr[r] = 0.f; }

    for (int j0 = 0; j0 < HW; j0 += 16) {
        // B = k^T: qk row-major with k at col offset 64 acts as transposed B
        v16bf bk0 = load_frag_b(qk + DQK, 128, b * HW + j0, 0, lane);
        v16bf bk1 = load_frag_b(qk + DQK, 128, b * HW + j0, 32, lane);
        v8f s = zero8();
        s = wmma_bf16(aq0, bk0, s);
        s = wmma_bf16(aq1, bk1, s);
        #pragma unroll
        for (int r = 0; r < 8; ++r) {
            float sv = s[r];
            float tm = sv;                    // max over the 16 j-lanes (half-wave)
            tm = fmaxf(tm, __shfl_xor(tm, 1));
            tm = fmaxf(tm, __shfl_xor(tm, 2));
            tm = fmaxf(tm, __shfl_xor(tm, 4));
            tm = fmaxf(tm, __shfl_xor(tm, 8));
            float mnew = fmaxf(mr[r], tm);
            float p = __expf(sv - mnew);
            float ts = p;
            ts += __shfl_xor(ts, 1);
            ts += __shfl_xor(ts, 2);
            ts += __shfl_xor(ts, 4);
            ts += __shfl_xor(ts, 8);
            lr[r] = lr[r] * __expf(mr[r] - mnew) + ts;
            mr[r] = mnew;
        }
    }
    if (m == 0) {
        #pragma unroll
        for (int r = 0; r < 8; ++r) {
            int i = i0 + 8 * hi + r;
            mrow[b * HW + i] = mr[r];
            linv[b * HW + i] = 1.0f / lr[r];
        }
    }
}

// ---------------- Kernel 4: out = gamma * (P^T V) + x (flash pass 2) ---------
// Block: batch b, 16 output rows j, 8 waves. Per outer iteration the block
// consumes an i-tile of 256: wave w computes S/P for its own 32-row i-chunk
// (no duplicated S work), publishes the bf16 P^T A-fragment to LDS (layout is
// lane-identical for all consumers -> raw b128 LDS copies), then every wave
// runs P^T·V WMMAs for its 64-channel c-range over all 8 chunks.
__global__ __launch_bounds__(256) void attn_out_kernel(const bf16_t* __restrict__ qk,
                                                       const bf16_t* __restrict__ vT,
                                                       const float* __restrict__ mrow,
                                                       const float* __restrict__ linv,
                                                       const float* __restrict__ x,
                                                       const float* __restrict__ gamma,
                                                       float* __restrict__ out) {
    __shared__ __align__(16) v8bf pshare[8][32][2];   // 8 KB: 8 chunks x 32 lanes x 32B

    int lane = threadIdx.x & 31;
    int w = threadIdx.x >> 5;
    int b  = blockIdx.x >> 8;
    int j0 = (blockIdx.x & 255) * 16;
    int m = lane & 15, hi = lane >> 4;
    int cbase = w * 64;

    // k^T fragments for this fixed j-tile (reused across the whole i loop)
    v16bf bk0 = load_frag_b(qk + DQK, 128, b * HW + j0, 0, lane);
    v16bf bk1 = load_frag_b(qk + DQK, 128, b * HW + j0, 32, lane);

    v8f acc[4];
    #pragma unroll
    for (int t = 0; t < 4; ++t) acc[t] = zero8();

    const float* mb = mrow + b * HW;
    const float* lb = linv + b * HW;
    const bf16_t* vTb = vT + (size_t)b * C_ * HW;

    for (int ii0 = 0; ii0 < HW; ii0 += 256) {
        int i0 = ii0 + w * 32;   // this wave's private i-chunk

        // S tiles for my chunk: 4 WMMAs (was 8 x redundant before)
        v8f s0 = zero8(), s1 = zero8();
        v16bf a00 = load_frag_a(qk, 128, b * HW + i0, 0, lane);
        v16bf a01 = load_frag_a(qk, 128, b * HW + i0, 32, lane);
        s0 = wmma_bf16(a00, bk0, s0);
        s0 = wmma_bf16(a01, bk1, s0);
        v16bf a10 = load_frag_a(qk, 128, b * HW + i0 + 16, 0, lane);
        v16bf a11 = load_frag_a(qk, 128, b * HW + i0 + 16, 32, lane);
        s1 = wmma_bf16(a10, bk0, s1);
        s1 = wmma_bf16(a11, bk1, s1);

        // P[i,j] = exp(s - m_i) / l_i ; D-tile layout == A-fragment of P^T
        BF16x16 pa;
        #pragma unroll
        for (int e = 0; e < 8; ++e) {
            int i_off = i0 + 8 * hi + e;
            float p0 = __expf(s0[e] - mb[i_off]) * lb[i_off];
            float p1 = __expf(s1[e] - mb[i_off + 16]) * lb[i_off + 16];
            pa.v[e]     = (bf16_t)p0;
            pa.v[e + 8] = (bf16_t)p1;
        }

        // prefetch next iteration's v^T tiles for this wave's c-range
        if (ii0 + 256 < HW) {
            #pragma unroll
            for (int t = 0; t < 4; ++t)
                __builtin_prefetch(vTb + (size_t)(cbase + t * 16 + m) * HW + ii0 + 256 + 16 * hi,
                                   0, 1);
        }

        __syncthreads();   // previous iteration's LDS reads complete
        pshare[w][lane][0] = pa.h[0];
        pshare[w][lane][1] = pa.h[1];
        __syncthreads();   // all chunks published

        // P^T · V over the full 256-row i-tile, K=32 per WMMA
        #pragma unroll
        for (int q = 0; q < 8; ++q) {
            BF16x16 paq;
            paq.h[0] = pshare[q][lane][0];
            paq.h[1] = pshare[q][lane][1];
            int iq = ii0 + q * 32;
            #pragma unroll
            for (int t = 0; t < 4; ++t) {
                v16bf bv = load_frag_b(vTb, HW, cbase + t * 16, iq, lane);
                acc[t] = wmma_bf16(paq.v, bv, acc[t]);
            }
        }
    }

    float g = gamma[0];
    #pragma unroll
    for (int t = 0; t < 4; ++t) {
        int c = cbase + t * 16 + m;
        #pragma unroll
        for (int r = 0; r < 8; ++r) {
            int j = j0 + r + 8 * hi;
            size_t idx = ((size_t)(b * HW + j)) * C_ + c;
            out[idx] = g * acc[t][r] + x[idx];
        }
    }
}

extern "C" void kernel_launch(void* const* d_in, const int* in_sizes, int n_in,
                              void* d_out, int out_size, void* d_ws, size_t ws_size,
                              hipStream_t stream) {
    (void)in_sizes; (void)n_in; (void)out_size; (void)ws_size;
    const float* x     = (const float*)d_in[0];
    const float* Wq    = (const float*)d_in[1];
    const float* bq    = (const float*)d_in[2];
    const float* Wk    = (const float*)d_in[3];
    const float* bk    = (const float*)d_in[4];
    const float* Wv    = (const float*)d_in[5];
    const float* bv    = (const float*)d_in[6];
    const float* gamma = (const float*)d_in[7];
    float* out = (float*)d_out;

    char* ws = (char*)d_ws;
    size_t off = 0;
    auto alloc = [&](size_t bytes) -> void* {
        void* p = ws + off;
        off = (off + bytes + 255) & ~(size_t)255;
        return p;
    };
    bf16_t* xb   = (bf16_t*)alloc((size_t)B_ * HW * C_ * 2);   // 32 MB
    bf16_t* wT   = (bf16_t*)alloc((size_t)NCOLS * C_ * 2);     // 640 KB
    float*  bc   = (float*)alloc((size_t)NCOLS * 4);
    bf16_t* qk   = (bf16_t*)alloc((size_t)B_ * HW * 128 * 2);  // 8 MB
    bf16_t* vT   = (bf16_t*)alloc((size_t)B_ * C_ * HW * 2);   // 32 MB
    float*  mrow = (float*)alloc((size_t)B_ * HW * 4);
    float*  liv  = (float*)alloc((size_t)B_ * HW * 4);

    int nconv = B_ * HW * C_;
    convert_pack_kernel<<<(nconv + 255) / 256, 256, 0, stream>>>(
        x, Wq, Wk, Wv, bq, bk, bv, xb, wT, bc);
    qkv_gemm_kernel<<<2560, 256, 0, stream>>>(xb, wT, bc, qk, vT);
    rowstats_kernel<<<256, 256, 0, stream>>>(qk, mrow, liv);
    attn_out_kernel<<<2048, 256, 0, stream>>>(qk, vT, mrow, liv, x, gamma, out);
}
